// MLDecoder_27633819582635
// MI455X (gfx1250) — matrix operand order repaired and checked by reading, not compiled, not run on hardware
//
#include <hip/hip_runtime.h>

// ---------------------------------------------------------------------------
// ML-Decoder head for MI455X (gfx1250).
// bf16 WMMA GEMMs everywhere; LDS tiles staged by the Tensor Data Mover
// (tensor_load_to_lds + s_wait_tensorcnt) when the builtin is available,
// with a plain vector-copy fallback otherwise.
// ---------------------------------------------------------------------------

typedef __bf16 bf16;
typedef __attribute__((ext_vector_type(16))) __bf16 v16bf;
typedef __attribute__((ext_vector_type(8)))  __bf16 v8bf;
typedef __attribute__((ext_vector_type(8)))  float  v8f;
typedef unsigned int u32x4 __attribute__((ext_vector_type(4)));
typedef int          i32x4 __attribute__((ext_vector_type(4)));
typedef int          i32x8 __attribute__((ext_vector_type(8)));

#ifndef USE_TDM
#if __has_builtin(__builtin_amdgcn_tensor_load_to_lds) && \
    __has_builtin(__builtin_amdgcn_s_wait_tensorcnt)
#define USE_TDM 1
#else
#define USE_TDM 0
#endif
#endif

#if USE_TDM
// 2D tile DMA: global (bf16 tensor, row length dim0_rem elems, row stride
// stride0 elems, dim1_rem rows remaining -> OOB rows/cols read as zero)
// into LDS at byte offset lds_off. Optional LDS padding of 16B per 64B row
// (matches LDA/LDB = 40 bf16 elems).
__device__ __forceinline__ void tdm_load_2d(
    const void* gaddr, unsigned lds_off, unsigned dim0_rem, unsigned dim1_rem,
    unsigned stride0, unsigned tile0, unsigned tile1, bool pad_rows) {
  unsigned long long ga = (unsigned long long)(uintptr_t)gaddr;
  u32x4 g0 = {};
  g0[0] = 1u;                                    // count=1, no gather
  g0[1] = lds_off;                               // LDS byte address
  g0[2] = (unsigned)(ga & 0xffffffffu);          // global_addr[31:0]
  g0[3] = (unsigned)((ga >> 32) & 0x1ffffffu) | (2u << 30);  // addr hi | type=2
  i32x8 g1 = {};
  unsigned w0 = (1u << 16);                      // data_size = 2 bytes
  if (pad_rows)                                  // +16B pad per 64B stored
    w0 |= (1u << 20) | (3u << 22) | (3u << 25);  // pad_en | interval=16DW | amount=4DW
  g1[0] = (int)w0;
  g1[1] = (int)((dim0_rem & 0xffffu) << 16);                       // dim0 lo
  g1[2] = (int)((dim0_rem >> 16) | ((dim1_rem & 0xffffu) << 16));  // dim0 hi | dim1 lo
  g1[3] = (int)((dim1_rem >> 16) | (tile0 << 16));                 // dim1 hi | tile0
  g1[4] = (int)(tile1 & 0xffffu);                                  // tile1 | tile2=0
  g1[5] = (int)stride0;                                            // dim0_stride lo
  g1[6] = 0;                                                       // stride hi | dim1_stride
  g1[7] = 0;
  i32x4 gz = {};
#if __clang_major__ >= 23
  i32x8 gz8 = {};
  __builtin_amdgcn_tensor_load_to_lds(g0, g1, gz, gz, gz8, 0);
#else
  __builtin_amdgcn_tensor_load_to_lds(g0, g1, gz, gz, 0);
#endif
}
#endif  // USE_TDM

// D = A(16x32) * B(32x16) + C, bf16 in, f32 accumulate
__device__ __forceinline__ v8f wmma_bf16(v16bf a, v16bf b, v8f c) {
  return __builtin_amdgcn_wmma_f32_16x16x32_bf16(false, a, false, b,
                                                 (short)0, c, false, false);
}

// A fragment (16x32) from LDS row-major [16][ld]; lane m=lane&15,
// elems 0..7 -> K=8*half+0..7, elems 8..15 -> K=16+8*half+0..7
__device__ __forceinline__ v16bf load_frag_a(const bf16* base, int ld) {
  const int lane = threadIdx.x & 31;
  const bf16* rp = base + (lane & 15) * ld + 8 * (lane >> 4);
  v8bf lo = *(const v8bf*)rp;
  v8bf hi = *(const v8bf*)(rp + 16);
  v16bf f;
#pragma unroll
  for (int i = 0; i < 8; ++i) { f[i] = lo[i]; f[8 + i] = hi[i]; }
  return f;
}

// B fragment (32x16) from LDS tile stored as B^T row-major [N][ld];
// lane n=lane&15, elems 0..7 -> K=8*half+0..7, elems 8..15 -> K=16+8*half+0..7
__device__ __forceinline__ v16bf load_frag_b(const bf16* base, int ld) {
  const int lane = threadIdx.x & 31;
  const bf16* rp = base + (lane & 15) * ld + 8 * (lane >> 4);
  v8bf lo = *(const v8bf*)rp;
  v8bf hi = *(const v8bf*)(rp + 16);
  v16bf f;
#pragma unroll
  for (int i = 0; i < 8; ++i) { f[i] = lo[i]; f[8 + i] = hi[i]; }
  return f;
}

__device__ __forceinline__ float block_sum(float v, float* sh) {
#pragma unroll
  for (int o = 16; o > 0; o >>= 1) v += __shfl_xor(v, o, 32);
  const int wave = threadIdx.x >> 5, lane = threadIdx.x & 31;
  if (lane == 0) sh[wave] = v;
  __syncthreads();
  float r = 0.f;
#pragma unroll
  for (int w = 0; w < 8; ++w) r += sh[w];
  __syncthreads();
  return r;
}

// ---------------------------------------------------------------------------
// x [B][2048][196] f32  ->  xt [B][196][2048] bf16  (tiled transpose+convert)
// ---------------------------------------------------------------------------
__global__ __launch_bounds__(256) void transpose_x_kernel(
    const float* __restrict__ x, bf16* __restrict__ xt) {
  __shared__ float tile[32][33];
  const int b = blockIdx.z;
  const int c0 = blockIdx.x * 32, s0 = blockIdx.y * 32;
  const int tx = threadIdx.x & 31, ty = threadIdx.x >> 5;
  for (int cc = ty; cc < 32; cc += 8) {
    int s = s0 + tx;
    tile[cc][tx] = (s < 196) ? x[((size_t)b * 2048 + c0 + cc) * 196 + s] : 0.f;
  }
  __syncthreads();
  for (int ss = ty; ss < 32; ss += 8) {
    int s = s0 + ss;
    if (s < 196)
      xt[((size_t)b * 196 + s) * 2048 + c0 + tx] = (bf16)tile[tx][ss];
  }
}

// W [K][N] f32 -> Wt [N][K] bf16   (K, N multiples of 32)
__global__ __launch_bounds__(256) void transpose_w_kernel(
    const float* __restrict__ w, bf16* __restrict__ wt, int K, int N) {
  __shared__ float tile[32][33];
  const int k0 = blockIdx.x * 32, n0 = blockIdx.y * 32;
  const int tx = threadIdx.x & 31, ty = threadIdx.x >> 5;
  for (int kk = ty; kk < 32; kk += 8)
    tile[kk][tx] = w[(size_t)(k0 + kk) * N + n0 + tx];
  __syncthreads();
  for (int nn = ty; nn < 32; nn += 8)
    wt[(size_t)(n0 + nn) * K + k0 + tx] = (bf16)tile[tx][nn];
}

// ---------------------------------------------------------------------------
// tgt = LayerNorm(2 * query_embed) once (batch-independent); f32 + bf16 copies
// ---------------------------------------------------------------------------
__global__ __launch_bounds__(256) void prep_tgt_kernel(
    const float* __restrict__ qe, const float* __restrict__ g1,
    const float* __restrict__ b1, float* __restrict__ tgt,
    bf16* __restrict__ tgtb) {
  __shared__ float sh[8];
  const int g = blockIdx.x, tid = threadIdx.x;
  float vals[3];
  float s1 = 0.f, s2 = 0.f;
#pragma unroll
  for (int j = 0; j < 3; ++j) {
    int i = tid + j * 256;
    float v = 2.f * qe[g * 768 + i];
    vals[j] = v; s1 += v; s2 += v * v;
  }
  float sum = block_sum(s1, sh);
  float sum2 = block_sum(s2, sh);
  float mean = sum * (1.f / 768.f);
  float rstd = rsqrtf(sum2 * (1.f / 768.f) - mean * mean + 1e-5f);
#pragma unroll
  for (int j = 0; j < 3; ++j) {
    int i = tid + j * 256;
    float v = (vals[j] - mean) * rstd * g1[i] + b1[i];
    tgt[g * 768 + i] = v;
    tgtb[g * 768 + i] = (bf16)v;
  }
}

// ---------------------------------------------------------------------------
// GEMM: C[M][768] = op(A[M][K] bf16 * Bt[768][K]^T bf16 + bias)
// BM=64 BN=256 BK=32 ; 256 threads = 8 waves (2x4), wave does 32x64 (2x4 wmma)
// EPI: 0 = relu->bf16, 1 = bf16, 2 = f32.   Requires M % 8 == 0.
// dynamic LDS: lds_a @0 (64*40 bf16 = 5120B), lds_b @5120 (256*40 = 20480B)
// ---------------------------------------------------------------------------
template <int EPI>
__global__ __launch_bounds__(256) void gemm_wmma_kernel(
    const bf16* __restrict__ A, const bf16* __restrict__ Bt,
    const float* __restrict__ bias, void* __restrict__ Cp, int M, int K) {
  constexpr int LDA = 40, LDB = 40;
  extern __shared__ char dsm[];
  bf16* lds_a = (bf16*)dsm;
  bf16* lds_b = (bf16*)(dsm + 5120);
  const int tid = threadIdx.x;
  const int wave = tid >> 5, lane = tid & 31;
  const int m0 = (wave >> 2) * 32, n0 = (wave & 3) * 64;
  const int blockM = blockIdx.y * 64;
  const int blockN = blockIdx.x * 256;
  v8f acc[2][4] = {};

  for (int k0 = 0; k0 < K; k0 += 32) {
#if USE_TDM
    if (wave == 0) {
      // A tile 64x32 (rows beyond M auto-zero), B tile 256x32
      tdm_load_2d(&A[(size_t)blockM * K + k0], 0u,
                  (unsigned)(K - k0), (unsigned)(M - blockM), (unsigned)K,
                  32u, 64u, true);
      tdm_load_2d(&Bt[(size_t)blockN * K + k0], 5120u,
                  (unsigned)(K - k0), 768u - (unsigned)blockN, (unsigned)K,
                  32u, 256u, true);
      __builtin_amdgcn_s_wait_tensorcnt(0);
    }
#else
    {  // A: 256 chunks of 8 bf16, 1 per thread
      int m = tid >> 2, kc = (tid & 3) * 8;
      int r = blockM + m;
      v8bf val = {};
      if (r < M) val = *(const v8bf*)&A[(size_t)r * K + k0 + kc];
      *(v8bf*)&lds_a[m * LDA + kc] = val;
    }
#pragma unroll
    for (int it = 0; it < 4; ++it) {  // B: 1024 chunks of 8, 4 per thread
      int c = it * 256 + tid;
      int n = c >> 2, kc = (c & 3) * 8;
      *(v8bf*)&lds_b[n * LDB + kc] =
          *(const v8bf*)&Bt[(size_t)(blockN + n) * K + k0 + kc];
    }
    if (k0 + 32 < K)
      __builtin_prefetch(&Bt[(size_t)(blockN + (tid >> 1)) * K + k0 + 32], 0, 1);
#endif
    __syncthreads();

    v16bf a0 = load_frag_a(lds_a + m0 * LDA, LDA);
    v16bf a1 = load_frag_a(lds_a + (m0 + 16) * LDA, LDA);
    v16bf bf[4];
#pragma unroll
    for (int j = 0; j < 4; ++j)
      bf[j] = load_frag_b(lds_b + (n0 + j * 16) * LDB, LDB);
#pragma unroll
    for (int j = 0; j < 4; ++j) {
      acc[0][j] = wmma_bf16(a0, bf[j], acc[0][j]);
      acc[1][j] = wmma_bf16(a1, bf[j], acc[1][j]);
    }
    __syncthreads();
  }

  // epilogue: C/D layout m = 8*(lane>>4)+r, n = lane&15; guard hoisted (M%8==0)
  const int ln = lane & 15;
  const int lmb = (lane >> 4) * 8;
#pragma unroll
  for (int mi = 0; mi < 2; ++mi) {
    int gmb = blockM + m0 + mi * 16 + lmb;
    if (gmb < M) {
#pragma unroll
      for (int ni = 0; ni < 4; ++ni) {
        int gn = blockN + n0 + ni * 16 + ln;
        float bv = bias[gn];
#pragma unroll
        for (int r = 0; r < 8; ++r) {
          float v = acc[mi][ni][r] + bv;
          if (EPI == 0) {
            v = fmaxf(v, 0.f);
            ((bf16*)Cp)[(size_t)(gmb + r) * 768 + gn] = (bf16)v;
          } else if (EPI == 1) {
            ((bf16*)Cp)[(size_t)(gmb + r) * 768 + gn] = (bf16)v;
          } else {
            ((float*)Cp)[(size_t)(gmb + r) * 768 + gn] = v;
          }
        }
      }
    }
  }
}

// ---------------------------------------------------------------------------
// Cross attention, one block per (b, h).  q bf16 [40][768] batch-independent.
// dynamic LDS (64512 B), phased overlays:
//   lq  @0      [48][96]  bf16      (phase A/B)
//   lk  @9216   [112][96] bf16      (phase A/B, two S halves)
//   lsc @30720  [40][196] f32       (scores)
//   lat @0      [48][224] bf16      (overlay, phase C+)
//   lvt @21504  [96][224] bf16      (overlay, phase D)
// ---------------------------------------------------------------------------
__global__ __launch_bounds__(256) void attn_kernel(
    const bf16* __restrict__ q, const bf16* __restrict__ kbuf,
    const bf16* __restrict__ vbuf, bf16* __restrict__ ctxb) {
  extern __shared__ char dsm[];
  bf16*  lq  = (bf16*)dsm;
  bf16*  lk  = (bf16*)(dsm + 9216);
  float* lsc = (float*)(dsm + 30720);
  bf16*  lat = (bf16*)dsm;
  bf16*  lvt = (bf16*)(dsm + 21504);

  const int tid = threadIdx.x, wave = tid >> 5, lane = tid & 31;
  const int bh = blockIdx.x, b = bh >> 3, h = bh & 7;
  const float scale = 0.1020620726159658f;  // 1/sqrt(96)

#if USE_TDM
  if (wave == 0)  // q tile: 40 rows (rows 40..47 auto-zero), 96 elems/row
    tdm_load_2d(q + h * 96, 0u, 96u, 40u, 768u, 96u, 48u, false);
#else
  for (int c = tid; c < 48 * 12; c += 256) {
    int g = c / 12, d8 = (c % 12) * 8;
    v8bf val = {};
    if (g < 40) val = *(const v8bf*)&q[(size_t)g * 768 + h * 96 + d8];
    *(v8bf*)&lq[g * 96 + d8] = val;
  }
#endif

  for (int halfS = 0; halfS < 2; ++halfS) {
    __syncthreads();
#if USE_TDM
    if (wave == 0) {
      tdm_load_2d(kbuf + ((size_t)b * 196 + halfS * 112) * 768 + h * 96, 9216u,
                  96u, (unsigned)(196 - halfS * 112), 768u, 96u, 112u, false);
      __builtin_amdgcn_s_wait_tensorcnt(0);
    }
#else
    for (int c = tid; c < 112 * 12; c += 256) {
      int s = c / 12, d8 = (c % 12) * 8;
      int gs = halfS * 112 + s;
      v8bf val = {};
      if (gs < 196)
        val = *(const v8bf*)&kbuf[((size_t)b * 196 + gs) * 768 + h * 96 + d8];
      *(v8bf*)&lk[s * 96 + d8] = val;
    }
#endif
    __syncthreads();
    for (int t = wave; t < 21; t += 8) {  // 3 m-tiles x 7 n-tiles
      int mt = t / 7, nt = t % 7;
      v8f acc = {};
#pragma unroll
      for (int kk = 0; kk < 3; ++kk) {  // K = 96
        v16bf fa = load_frag_a(lq + (mt * 16) * 96 + kk * 32, 96);
        v16bf fb = load_frag_b(lk + (nt * 16) * 96 + kk * 32, 96);
        acc = wmma_bf16(fa, fb, acc);
      }
      int mb = mt * 16 + (lane >> 4) * 8;
      int gs = halfS * 112 + nt * 16 + (lane & 15);
      if (mb < 40 && gs < 196) {
#pragma unroll
        for (int r = 0; r < 8; ++r) lsc[(mb + r) * 196 + gs] = acc[r] * scale;
      }
    }
  }
  __syncthreads();

  // zero attn buffer (covers all padding), then row softmax
  for (int c = tid; c < 48 * 224 / 8; c += 256) ((v8bf*)lat)[c] = (v8bf){};
  __syncthreads();
  for (int g = wave; g < 40; g += 8) {
    float mx = -1e30f;
    for (int s = lane; s < 196; s += 32) mx = fmaxf(mx, lsc[g * 196 + s]);
#pragma unroll
    for (int o = 16; o > 0; o >>= 1) mx = fmaxf(mx, __shfl_xor(mx, o, 32));
    float sum = 0.f;
    for (int s = lane; s < 196; s += 32) {
      float e = __expf(lsc[g * 196 + s] - mx);
      lsc[g * 196 + s] = e;
      sum += e;
    }
#pragma unroll
    for (int o = 16; o > 0; o >>= 1) sum += __shfl_xor(sum, o, 32);
    float inv = 1.f / sum;
    for (int s = lane; s < 196; s += 32)
      lat[g * 224 + s] = (bf16)(lsc[g * 196 + s] * inv);
  }
  __syncthreads();

  // stage v^T [d][s] (manual: transpose)
  for (int c = tid; c < 12 * 224; c += 256) {
    int d8 = (c % 12) * 8, s = c / 12;
    v8bf val = {};
    if (s < 196)
      val = *(const v8bf*)&vbuf[((size_t)b * 196 + s) * 768 + h * 96 + d8];
#pragma unroll
    for (int j = 0; j < 8; ++j) lvt[(d8 + j) * 224 + s] = val[j];
  }
  __syncthreads();

  // ctx = attn * v : 3 m-tiles x 6 n-tiles, K = 224; bf16 out
  for (int t = wave; t < 18; t += 8) {
    int mt = t / 6, nt = t % 6;
    v8f acc = {};
#pragma unroll
    for (int kk = 0; kk < 7; ++kk) {
      v16bf fa = load_frag_a(lat + (mt * 16) * 224 + kk * 32, 224);
      v16bf fb = load_frag_b(lvt + (nt * 16) * 224 + kk * 32, 224);
      acc = wmma_bf16(fa, fb, acc);
    }
    int mb = mt * 16 + (lane >> 4) * 8;
    if (mb < 40) {
      int n = nt * 16 + (lane & 15);
#pragma unroll
      for (int r = 0; r < 8; ++r)
        ctxb[((size_t)b * 40 + mb + r) * 768 + h * 96 + n] = (bf16)acc[r];
    }
  }
}

// ---------------------------------------------------------------------------
// Tail: t2 = LN2(tgt + y) ; ff = relu(t2 W1 + b1) W2 + b2 ; h = LN3(t2 + ff);
// logits[b, g*25+f] = h . dup_W[g,:,f] + dup_b.   One block per (b, g).
// ---------------------------------------------------------------------------
__global__ __launch_bounds__(256) void tail_kernel(
    const float* __restrict__ y, const float* __restrict__ tgt,
    const float* __restrict__ ln2g, const float* __restrict__ ln2b,
    const float* __restrict__ W1, const float* __restrict__ b1,
    const float* __restrict__ W2, const float* __restrict__ b2,
    const float* __restrict__ ln3g, const float* __restrict__ ln3b,
    const float* __restrict__ dupW, const float* __restrict__ dupb,
    float* __restrict__ out) {
  __shared__ float t2[768];
  __shared__ float ff[80];
  __shared__ float hbuf[768];
  __shared__ float sh[8];
  const int tid = threadIdx.x, wave = tid >> 5, lane = tid & 31;
  const int row = blockIdx.x, b = row / 40, g = row % 40;

  float s1 = 0.f, s2 = 0.f;
  for (int i = tid; i < 768; i += 256) {
    float v = tgt[g * 768 + i] + y[(size_t)row * 768 + i];
    t2[i] = v; s1 += v; s2 += v * v;
  }
  float sum = block_sum(s1, sh), sum2 = block_sum(s2, sh);
  float mean = sum * (1.f / 768.f);
  float rstd = rsqrtf(sum2 * (1.f / 768.f) - mean * mean + 1e-5f);
  for (int i = tid; i < 768; i += 256)
    t2[i] = (t2[i] - mean) * rstd * ln2g[i] + ln2b[i];
  __syncthreads();

  for (int f = wave; f < 80; f += 8) {
    float a = 0.f;
    for (int e = lane; e < 768; e += 32) a += t2[e] * W1[e * 80 + f];
#pragma unroll
    for (int o = 16; o > 0; o >>= 1) a += __shfl_xor(a, o, 32);
    if (lane == 0) { a += b1[f]; ff[f] = a > 0.f ? a : 0.f; }
  }
  __syncthreads();

  s1 = s2 = 0.f;
  for (int i = tid; i < 768; i += 256) {
    float a = b2[i];
    for (int e = 0; e < 80; ++e) a += ff[e] * W2[e * 768 + i];
    float v = t2[i] + a;
    hbuf[i] = v; s1 += v; s2 += v * v;
  }
  sum = block_sum(s1, sh); sum2 = block_sum(s2, sh);
  mean = sum * (1.f / 768.f);
  rstd = rsqrtf(sum2 * (1.f / 768.f) - mean * mean + 1e-5f);
  for (int i = tid; i < 768; i += 256)
    hbuf[i] = (hbuf[i] - mean) * rstd * ln3g[i] + ln3b[i];
  __syncthreads();

  for (int f = wave; f < 25; f += 8) {
    float a = 0.f;
    for (int e = lane; e < 768; e += 32)
      a += hbuf[e] * dupW[((size_t)g * 768 + e) * 25 + f];
#pragma unroll
    for (int o = 16; o > 0; o >>= 1) a += __shfl_xor(a, o, 32);
    if (lane == 0) out[(size_t)b * 1000 + g * 25 + f] = a + dupb[g * 25 + f];
  }
}

// ---------------------------------------------------------------------------
extern "C" void kernel_launch(void* const* d_in, const int* in_sizes, int n_in,
                              void* d_out, int out_size, void* d_ws,
                              size_t ws_size, hipStream_t stream) {
  (void)in_sizes; (void)n_in; (void)out_size; (void)ws_size;
  const float* x   = (const float*)d_in[0];
  const float* eW  = (const float*)d_in[1];
  const float* eb  = (const float*)d_in[2];
  const float* qe  = (const float*)d_in[3];
  const float* Wq  = (const float*)d_in[4];
  const float* bq  = (const float*)d_in[5];
  const float* Wk  = (const float*)d_in[6];
  const float* bk  = (const float*)d_in[7];
  const float* Wv  = (const float*)d_in[8];
  const float* bv  = (const float*)d_in[9];
  const float* Wo  = (const float*)d_in[10];
  const float* bo  = (const float*)d_in[11];
  const float* W1  = (const float*)d_in[12];
  const float* b1  = (const float*)d_in[13];
  const float* W2  = (const float*)d_in[14];
  const float* b2  = (const float*)d_in[15];
  const float* l1g = (const float*)d_in[16];
  const float* l1b = (const float*)d_in[17];
  const float* l2g = (const float*)d_in[18];
  const float* l2b = (const float*)d_in[19];
  const float* l3g = (const float*)d_in[20];
  const float* l3b = (const float*)d_in[21];
  const float* dW  = (const float*)d_in[22];
  const float* db  = (const float*)d_in[23];
  float* out = (float*)d_out;

  // workspace layout (256B aligned blocks)
  char* w = (char*)d_ws;
  size_t o = 0;
  auto alloc = [&](size_t bytes) -> void* {
    void* p = w + o;
    o += (bytes + 255) & ~(size_t)255;
    return p;
  };
  float* tgt  = (float*)alloc((size_t)40 * 768 * 4);
  bf16*  tgtb = (bf16*)alloc((size_t)40 * 768 * 2);
  bf16*  qb   = (bf16*)alloc((size_t)40 * 768 * 2);
  bf16*  xt   = (bf16*)alloc((size_t)25088 * 2048 * 2);
  bf16*  mem  = (bf16*)alloc((size_t)25088 * 768 * 2);
  bf16*  kb   = (bf16*)alloc((size_t)25088 * 768 * 2);
  bf16*  vb   = (bf16*)alloc((size_t)25088 * 768 * 2);
  bf16*  ctxb = (bf16*)alloc((size_t)5120 * 768 * 2);
  float* yb   = (float*)alloc((size_t)5120 * 768 * 4);
  bf16*  eWt  = (bf16*)alloc((size_t)768 * 2048 * 2);
  bf16*  Wqt  = (bf16*)alloc((size_t)768 * 768 * 2);
  bf16*  Wkt  = (bf16*)alloc((size_t)768 * 768 * 2);
  bf16*  Wvt  = (bf16*)alloc((size_t)768 * 768 * 2);
  bf16*  Wot  = (bf16*)alloc((size_t)768 * 768 * 2);

  // one-time input transforms (bf16, weights pre-transposed to [N][K])
  transpose_x_kernel<<<dim3(64, 7, 128), 256, 0, stream>>>(x, xt);
  transpose_w_kernel<<<dim3(64, 24), 256, 0, stream>>>(eW, eWt, 2048, 768);
  transpose_w_kernel<<<dim3(24, 24), 256, 0, stream>>>(Wq, Wqt, 768, 768);
  transpose_w_kernel<<<dim3(24, 24), 256, 0, stream>>>(Wk, Wkt, 768, 768);
  transpose_w_kernel<<<dim3(24, 24), 256, 0, stream>>>(Wv, Wvt, 768, 768);
  transpose_w_kernel<<<dim3(24, 24), 256, 0, stream>>>(Wo, Wot, 768, 768);
  prep_tgt_kernel<<<40, 256, 0, stream>>>(qe, l1g, l1b, tgt, tgtb);

  const unsigned GEMM_LDS = (64 * 40 + 256 * 40) * 2;  // 25600 B
  // mem = relu(xs @ embed_W + b)
  gemm_wmma_kernel<0><<<dim3(3, 392), 256, GEMM_LDS, stream>>>(xt, eWt, eb, mem, 25088, 2048);
  // q = tgt @ Wq + bq
  gemm_wmma_kernel<1><<<dim3(3, 1), 256, GEMM_LDS, stream>>>(tgtb, Wqt, bq, qb, 40, 768);
  // k, v
  gemm_wmma_kernel<1><<<dim3(3, 392), 256, GEMM_LDS, stream>>>(mem, Wkt, bk, kb, 25088, 768);
  gemm_wmma_kernel<1><<<dim3(3, 392), 256, GEMM_LDS, stream>>>(mem, Wvt, bv, vb, 25088, 768);
  // attention per (b, h)
  attn_kernel<<<128 * 8, 256, 64512, stream>>>(qb, kb, vb, ctxb);
  // y = ctx @ Wo + bo
  gemm_wmma_kernel<2><<<dim3(3, 80), 256, GEMM_LDS, stream>>>(ctxb, Wot, bo, yb, 5120, 768);
  // LN2 + FF + LN3 + GroupFC
  tail_kernel<<<5120, 256, 0, stream>>>(yb, tgt, l2g, l2b, W1, b1, W2, b2,
                                        l3g, l3b, dW, db, out);
}
